// SelfAttention_28226525069905
// MI455X (gfx1250) — matrix operand order repaired and checked by reading, not compiled, not run on hardware
//
#include <hip/hip_runtime.h>

#define D_MODEL 1024
#define NHEAD   16
#define DHEAD   64
#define RANK    32
#define SEQ     2048
#define BATCH   2
#define NTOK    (BATCH * SEQ)   // 4096
#define NPAD    72              // LDS row stride (bf16 elems), 144B, 16B-aligned

typedef __attribute__((ext_vector_type(16))) __bf16 bf16x16;
typedef __attribute__((ext_vector_type(8)))  float  f32x8;
typedef __attribute__((ext_vector_type(4)))  float  f32x4;

union Frag {
    bf16x16 v;
    f32x4   h[2];
};

static __device__ __forceinline__ f32x8 wmma_bf16(const Frag& a, const Frag& b, f32x8 c) {
    return __builtin_amdgcn_wmma_f32_16x16x32_bf16(
        false, a.v, false, b.v, (short)0, c, false, false);
}

// ---------------------------------------------------------------------------
// Stage 0a: fp32 -> bf16 convert
// ---------------------------------------------------------------------------
__global__ void cvt_f32_bf16(const float* __restrict__ in, __bf16* __restrict__ out, int n) {
    int i = blockIdx.x * 256 + threadIdx.x;
    if (i < n) out[i] = (__bf16)in[i];
}

// ---------------------------------------------------------------------------
// Stage 0b: Wm = W + dB @ dW  (rank-32), store bf16 row-major (out_f, in_f)
// ---------------------------------------------------------------------------
__global__ void merge_lora(const float* __restrict__ W, const float* __restrict__ dW,
                           const float* __restrict__ dB, __bf16* __restrict__ out) {
    int idx = blockIdx.x * 256 + threadIdx.x;       // 0 .. 1024*1024-1
    int i = idx >> 10;                              // output feature (row of Wm)
    int j = idx & 1023;                             // input feature
    float acc = W[idx];
#pragma unroll
    for (int r = 0; r < RANK; ++r)
        acc += dB[i * RANK + r] * dW[r * D_MODEL + j];
    out[idx] = (__bf16)acc;
}

// ---------------------------------------------------------------------------
// Stage 1/3: Y = X @ Wm^T + b.  M=4096, N=K=1024, bf16 in, fp32 accum.
// Block: 128 threads (4 waves). Block tile 128(M) x 64(N); wave tile 32 x 64.
// ---------------------------------------------------------------------------
template <bool F32OUT>
__global__ __launch_bounds__(128)
void lora_gemm(const __bf16* __restrict__ X, const __bf16* __restrict__ Wm,
               const float* __restrict__ bias, void* __restrict__ Yv) {
    const int lane = threadIdx.x & 31;
    const int wave = threadIdx.x >> 5;
    const int half = lane >> 4;          // 0: lanes 0-15, 1: lanes 16-31
    const int l16  = lane & 15;
    const int mbase = blockIdx.x * 128 + wave * 32;
    const int nbase = blockIdx.y * 64;

    f32x8 acc[2][4] = {};

    const __bf16* arow0 = X + (size_t)(mbase + l16) * D_MODEL + half * 8;
    const __bf16* arow1 = arow0 + 16 * D_MODEL;
    const __bf16* brow  = Wm + (size_t)(nbase + l16) * D_MODEL + half * 8;

    for (int k0 = 0; k0 < D_MODEL; k0 += 32) {
        Frag a0, a1;
        a0.h[0] = *(const f32x4*)(arow0 + k0);        // K = k0 + half*8 .. +7
        a0.h[1] = *(const f32x4*)(arow0 + k0 + 16);   // K = k0+16+half*8 .. +7
        a1.h[0] = *(const f32x4*)(arow1 + k0);
        a1.h[1] = *(const f32x4*)(arow1 + k0 + 16);
#pragma unroll
        for (int s = 0; s < 4; ++s) {
            Frag b;
            const __bf16* bp = brow + (size_t)s * 16 * D_MODEL + k0;
            b.h[0] = *(const f32x4*)(bp);             // K = k0+half*8 .. +7
            b.h[1] = *(const f32x4*)(bp + 16);
            acc[0][s] = wmma_bf16(a0, b, acc[0][s]);
            acc[1][s] = wmma_bf16(a1, b, acc[1][s]);
        }
    }

#pragma unroll
    for (int mi = 0; mi < 2; ++mi) {
#pragma unroll
        for (int s = 0; s < 4; ++s) {
            int col = nbase + s * 16 + l16;
            float bv = bias[col];
            int row0 = mbase + mi * 16 + half * 8;
#pragma unroll
            for (int r = 0; r < 8; ++r) {
                float v = acc[mi][s][r] + bv;
                if constexpr (F32OUT)
                    ((float*)Yv)[(size_t)(row0 + r) * D_MODEL + col] = v;
                else
                    ((__bf16*)Yv)[(size_t)(row0 + r) * D_MODEL + col] = (__bf16)v;
            }
        }
    }
}

// ---------------------------------------------------------------------------
// Stage 2: flash attention. grid = (32 q-tiles, 32 b*h). Block 128 thr / 4 waves.
// K tile staged to LDS with async global->LDS DMA (ASYNCcnt); V tile staged
// transposed (vector global load + scalar LDS stores). bf16 WMMA throughout.
// ---------------------------------------------------------------------------
__global__ __launch_bounds__(128)
void flash_attn(const __bf16* __restrict__ Q, const __bf16* __restrict__ K,
                const __bf16* __restrict__ V, __bf16* __restrict__ O) {
    __shared__ __align__(16) __bf16 sK[64 * NPAD];         // K tile: [key][d]
    __shared__ __align__(16) __bf16 sV[DHEAD * NPAD];      // V^T tile: [d][key]
    __shared__ __align__(16) __bf16 sP[4][16 * NPAD];      // per-wave P: [q][key]

    const int lane = threadIdx.x & 31;
    const int wave = threadIdx.x >> 5;
    const int half = lane >> 4;
    const int l16  = lane & 15;
    const int bh = blockIdx.y;
    const int b  = bh >> 4;
    const int h  = bh & 15;
    const int qbase = blockIdx.x * 64 + wave * 16;
    const size_t headoff = (size_t)h * DHEAD;

    // Q fragments (held across the whole key loop): d chunks 0-31, 32-63
    Frag aq[2];
    {
        const __bf16* qp = Q + ((size_t)(b * SEQ + qbase + l16)) * D_MODEL + headoff + half * 8;
#pragma unroll
        for (int c = 0; c < 2; ++c) {
            aq[c].h[0] = *(const f32x4*)(qp + c * 32);
            aq[c].h[1] = *(const f32x4*)(qp + c * 32 + 16);
        }
    }

    float mI[8], lI[8];
#pragma unroll
    for (int r = 0; r < 8; ++r) { mI[r] = -1e30f; lI[r] = 0.f; }
    f32x8 o[4] = {};

    for (int kt = 0; kt < SEQ / 64; ++kt) {
        __syncthreads();
        // Cooperative staging: 64 keys x 64 d, 128 threads, 4 x 16B chunks each.
        //  K: row-major [key][d] via async global->LDS DMA.
        //  V: transposed [d][key] via b128 load + 8 scalar LDS stores.
#pragma unroll
        for (int i = 0; i < 4; ++i) {
            int idx8 = i * 128 + threadIdx.x;        // 0..511 chunks of 8 bf16
            int key  = idx8 >> 3;
            int d0   = (idx8 & 7) * 8;
            size_t goff = (size_t)(b * SEQ + kt * 64 + key) * D_MODEL + headoff + d0;

            unsigned kdst = (unsigned)(size_t)&sK[key * NPAD + d0];
            unsigned long long kga = (unsigned long long)(K + goff);
            asm volatile("global_load_async_to_lds_b128 %0, %1, off"
                         :: "v"(kdst), "v"(kga) : "memory");

            union { f32x4 v4; __bf16 e[8]; } t;
            t.v4 = *(const f32x4*)(V + goff);
#pragma unroll
            for (int j = 0; j < 8; ++j)
                sV[(d0 + j) * NPAD + key] = t.e[j];
        }
        asm volatile("s_wait_asynccnt 0" ::: "memory");
        __syncthreads();

        // S = Q K^T  (16 q x 64 keys per wave), K fragments from LDS
        f32x8 s[4] = {};
#pragma unroll
        for (int sub = 0; sub < 4; ++sub) {
            const __bf16* kp = sK + (sub * 16 + l16) * NPAD + half * 8;
#pragma unroll
            for (int c = 0; c < 2; ++c) {
                Frag bk;
                bk.h[0] = *(const f32x4*)(kp + c * 32);
                bk.h[1] = *(const f32x4*)(kp + c * 32 + 16);
                s[sub] = wmma_bf16(aq[c], bk, s[sub]);
            }
        }

        // Online softmax over the 64-key slab. Row r of this wave lives in
        // lanes [half*16, half*16+16); shfl_xor masks < 16 stay in that half.
        // Reductions are step-major: 8 independent bpermutes per butterfly
        // step so the LDS-permute latency pipelines across rows.
        float mx[8], rs[8], alpha[8];
#pragma unroll
        for (int r = 0; r < 8; ++r) {
            float s0 = s[0][r] * 0.125f, s1 = s[1][r] * 0.125f;
            float s2 = s[2][r] * 0.125f, s3 = s[3][r] * 0.125f;
            s[0][r] = s0; s[1][r] = s1; s[2][r] = s2; s[3][r] = s3;
            mx[r] = fmaxf(fmaxf(s0, s1), fmaxf(s2, s3));
        }
#pragma unroll
        for (int off = 1; off < 16; off <<= 1)
#pragma unroll
            for (int r = 0; r < 8; ++r)
                mx[r] = fmaxf(mx[r], __shfl_xor(mx[r], off, 32));

#pragma unroll
        for (int r = 0; r < 8; ++r) {
            float mn = fmaxf(mI[r], mx[r]);
            alpha[r] = __expf(mI[r] - mn);
            mI[r] = mn;
            float acc = 0.f;
#pragma unroll
            for (int sub = 0; sub < 4; ++sub) {
                float p = __expf(s[sub][r] - mn);
                s[sub][r] = p;
                acc += p;
            }
            rs[r] = acc;
        }
#pragma unroll
        for (int off = 1; off < 16; off <<= 1)
#pragma unroll
            for (int r = 0; r < 8; ++r)
                rs[r] += __shfl_xor(rs[r], off, 32);
#pragma unroll
        for (int r = 0; r < 8; ++r)
            lI[r] = lI[r] * alpha[r] + rs[r];

#pragma unroll
        for (int sub = 0; sub < 4; ++sub)
#pragma unroll
            for (int r = 0; r < 8; ++r)
                o[sub][r] *= alpha[r];

        // Re-layout P: C-format -> LDS [q][key] -> A-format fragments
        __bf16* Pw = &sP[wave][0];
#pragma unroll
        for (int sub = 0; sub < 4; ++sub)
#pragma unroll
            for (int r = 0; r < 8; ++r)
                Pw[(r + half * 8) * NPAD + sub * 16 + l16] = (__bf16)s[sub][r];

        // O += P V  (K-dim = keys, two chunks of 32)
#pragma unroll
        for (int c = 0; c < 2; ++c) {
            Frag ap;
            const __bf16* pr = Pw + l16 * NPAD + c * 32 + half * 8;
            ap.h[0] = *(const f32x4*)(pr);
            ap.h[1] = *(const f32x4*)(pr + 16);
#pragma unroll
            for (int sub = 0; sub < 4; ++sub) {
                Frag bv;
                const __bf16* vr = sV + (sub * 16 + l16) * NPAD + c * 32 + half * 8;
                bv.h[0] = *(const f32x4*)(vr);
                bv.h[1] = *(const f32x4*)(vr + 16);
                o[sub] = wmma_bf16(ap, bv, o[sub]);
            }
        }
    }

    // Normalize and store (b, q, h, d) as bf16
#pragma unroll
    for (int sub = 0; sub < 4; ++sub) {
#pragma unroll
        for (int r = 0; r < 8; ++r) {
            float v = o[sub][r] / lI[r];
            size_t tq = (size_t)(b * SEQ + qbase + half * 8 + r);
            O[tq * D_MODEL + headoff + sub * 16 + l16] = (__bf16)v;
        }
    }
}

// ---------------------------------------------------------------------------
extern "C" void kernel_launch(void* const* d_in, const int* in_sizes, int n_in,
                              void* d_out, int out_size, void* d_ws, size_t ws_size,
                              hipStream_t stream) {
    const float* x = (const float*)d_in[0];
    const float* W[4]   = {(const float*)d_in[1],  (const float*)d_in[5],
                           (const float*)d_in[9],  (const float*)d_in[13]};
    const float* bia[4] = {(const float*)d_in[2],  (const float*)d_in[6],
                           (const float*)d_in[10], (const float*)d_in[14]};
    const float* dW[4]  = {(const float*)d_in[3],  (const float*)d_in[7],
                           (const float*)d_in[11], (const float*)d_in[15]};
    const float* dB[4]  = {(const float*)d_in[4],  (const float*)d_in[8],
                           (const float*)d_in[12], (const float*)d_in[16]};

    char* ws = (char*)d_ws;
    const size_t MB = 1024 * 1024;
    __bf16* xb = (__bf16*)(ws);                 // 8 MiB: 4096x1024 bf16
    __bf16* wm[4];
    for (int i = 0; i < 4; ++i) wm[i] = (__bf16*)(ws + 8 * MB + (size_t)i * 2 * MB);
    __bf16* qb = (__bf16*)(ws + 16 * MB);       // 8 MiB each
    __bf16* kb = (__bf16*)(ws + 24 * MB);
    __bf16* vb = (__bf16*)(ws + 32 * MB);
    __bf16* ab = (__bf16*)(ws + 40 * MB);       // attention output (bf16)

    // Stage 0: converts + LoRA merges
    cvt_f32_bf16<<<(NTOK * D_MODEL + 255) / 256, 256, 0, stream>>>(x, xb, NTOK * D_MODEL);
    for (int i = 0; i < 4; ++i)
        merge_lora<<<(D_MODEL * D_MODEL) / 256, 256, 0, stream>>>(W[i], dW[i], dB[i], wm[i]);

    // Stage 1: Q, K, V projections
    dim3 ggrid(NTOK / 128, D_MODEL / 64);
    lora_gemm<false><<<ggrid, 128, 0, stream>>>(xb, wm[0], bia[0], qb);
    lora_gemm<false><<<ggrid, 128, 0, stream>>>(xb, wm[1], bia[1], kb);
    lora_gemm<false><<<ggrid, 128, 0, stream>>>(xb, wm[2], bia[2], vb);

    // Stage 2: attention
    dim3 agrid(SEQ / 64, BATCH * NHEAD);
    flash_attn<<<agrid, 128, 0, stream>>>(qb, kb, vb, ab);

    // Stage 3: output projection -> fp32 d_out
    lora_gemm<true><<<ggrid, 128, 0, stream>>>(ab, wm[3], bia[3], (float*)d_out);
}
